// GraphAttentionLayer_83365315215877
// MI455X (gfx1250) — matrix operand order repaired and checked by reading, not compiled, not run on hardware
//
#include <hip/hip_runtime.h>
#include <hip/hip_bf16.h>

#define NN   8192
#define DIN  128
#define DOUT 128

typedef __attribute__((ext_vector_type(16))) _Float16 v16h;
typedef __attribute__((ext_vector_type(8)))  _Float16 v8h;
typedef __attribute__((ext_vector_type(8)))  float    v8f;

#define LOG2E 1.4426950408889634f
// mask value pre-scaled into log2 domain (-9e15 * log2e)
#define NEGL2 (-1.2984292e16f)

// ---------------------------------------------------------------------------
// Kernel 1: h = input @ W (fp32), hT (fp16, transposed [DOUT][NN]),
//           f1 = (h @ a1) * log2e, f2 = (h @ a2) * log2e  (log2-domain scores
//           so kernel 2 can use raw v_exp_f32 with no per-element multiply).
// ---------------------------------------------------------------------------
__global__ __launch_bounds__(128)
void gat_proj(const float* __restrict__ input, const float* __restrict__ W,
              const float* __restrict__ a, _Float16* __restrict__ hT,
              float* __restrict__ f1, float* __restrict__ f2)
{
    __shared__ float inRow[DIN];
    __shared__ float red[8];
    const int t   = threadIdx.x;          // output feature 0..127
    const float a1t = a[t];
    const float a2t = a[DOUT + t];
    const int iBase = blockIdx.x * 32;

    for (int r = 0; r < 32; ++r) {
        const int i = iBase + r;
        __syncthreads();
        inRow[t] = input[(size_t)i * DIN + t];
        __syncthreads();

        float hv = 0.f;
        #pragma unroll 16
        for (int k = 0; k < DIN; ++k)
            hv = __builtin_fmaf(inRow[k], W[k * DOUT + t], hv);

        hT[(size_t)t * NN + i] = (_Float16)hv;   // transposed fp16 copy

        float p1 = hv * a1t, p2 = hv * a2t;      // wave32 reduction
        #pragma unroll
        for (int off = 16; off > 0; off >>= 1) {
            p1 += __shfl_xor(p1, off, 32);
            p2 += __shfl_xor(p2, off, 32);
        }
        if ((t & 31) == 0) { red[t >> 5] = p1; red[4 + (t >> 5)] = p2; }
        __syncthreads();
        if (t == 0) {
            f1[i] = (red[0] + red[1] + red[2] + red[3]) * LOG2E;
            f2[i] = (red[4] + red[5] + red[6] + red[7]) * LOG2E;
        }
    }
}

// ---------------------------------------------------------------------------
// Kernel 2: fused mask + leakyrelu + online softmax + (attention @ h) + ELU.
// One wave32 per 16-row tile; fp16 operands, fp32 accumulation via
// v_wmma_f32_16x16x32_f16. Scores kept in log2 domain; raw v_exp_f32.
// Accumulator rescale skipped via wave-uniform branch when no row max moved.
// ---------------------------------------------------------------------------
__global__ __launch_bounds__(128)
void gat_attn(const int* __restrict__ adj, const _Float16* __restrict__ hT,
              const float* __restrict__ f1, const float* __restrict__ f2,
              float* __restrict__ out)
{
    const int lane  = threadIdx.x & 31;
    const int wave  = threadIdx.x >> 5;
    const int row16 = lane & 15;              // A-operand row / D-operand col
    const int kbase = (lane >> 4) << 3;       // A K-group base: 0 or 8
    const int hb    = (lane >> 4) << 4;       // unused; B also uses kbase (0 or 8)
    const int i0    = (blockIdx.x * 4 + wave) * 16;
    const int rowG  = i0 + row16;
    const float f1r = f1[rowG];               // already * log2e
    const int* __restrict__ adjRow = adj + (size_t)rowG * NN;

    v8f acc[8];
    #pragma unroll
    for (int n = 0; n < 8; ++n)
        acc[n] = (v8f){0.f,0.f,0.f,0.f,0.f,0.f,0.f,0.f};
    float m = -__builtin_inff();
    float l = 0.f;

    for (int j0 = 0; j0 < NN; j0 += 32) {
        __builtin_prefetch(adjRow + j0 + 64, 0, 1);   // global_prefetch_b8

        // 16 adj ints + 16 f2 floats per lane, in A-operand K order:
        // t<8 -> K=kbase+t ; t>=8 -> K=16+kbase+(t-8)
        int4 A0 = *(const int4*)(adjRow + j0 + kbase);
        int4 A1 = *(const int4*)(adjRow + j0 + kbase + 4);
        int4 A2 = *(const int4*)(adjRow + j0 + 16 + kbase);
        int4 A3 = *(const int4*)(adjRow + j0 + 16 + kbase + 4);
        float4 G0 = *(const float4*)(f2 + j0 + kbase);
        float4 G1 = *(const float4*)(f2 + j0 + kbase + 4);
        float4 G2 = *(const float4*)(f2 + j0 + 16 + kbase);
        float4 G3 = *(const float4*)(f2 + j0 + 16 + kbase + 4);

        const int   am[16] = {A0.x,A0.y,A0.z,A0.w, A1.x,A1.y,A1.z,A1.w,
                              A2.x,A2.y,A2.z,A2.w, A3.x,A3.y,A3.z,A3.w};
        const float fv[16] = {G0.x,G0.y,G0.z,G0.w, G1.x,G1.y,G1.z,G1.w,
                              G2.x,G2.y,G2.z,G2.w, G3.x,G3.y,G3.z,G3.w};

        float e[16];
        float em = -__builtin_inff();
        #pragma unroll
        for (int t = 0; t < 16; ++t) {
            float s = f1r + fv[t];
            s = fmaxf(s, 0.2f * s);                   // leaky relu (s scaled>0)
            s = am[t] > 0 ? s : NEGL2;                // adjacency mask
            e[t] = s;
            em = fmaxf(em, s);
        }
        em = fmaxf(em, __shfl_xor(em, 16, 32));       // combine row halves
        const float mnew  = fmaxf(m, em);
        const float scale = __builtin_amdgcn_exp2f(m - mnew);  // raw v_exp_f32

        float psum = 0.f;
        v16h Ah;
        #pragma unroll
        for (int t = 0; t < 16; ++t) {
            float p = __builtin_amdgcn_exp2f(e[t] - mnew);
            psum += p;
            Ah[t] = (_Float16)p;
        }
        psum += __shfl_xor(psum, 16, 32);
        l = l * scale + psum;
        m = mnew;

        // rescale accumulators only if some row's max moved (wave-uniform
        // branch: s_cbranch, EXEC stays all-ones for the WMMAs below)
        if (__ballot(scale != 1.0f) != 0ull) {
            float rsc[8];
            #pragma unroll
            for (int k = 0; k < 8; ++k) rsc[k] = __shfl(scale, kbase + k, 32);
            #pragma unroll
            for (int n = 0; n < 8; ++n)
                #pragma unroll
                for (int k = 0; k < 8; ++k) acc[n][k] *= rsc[k];
        }

        // 8 N-tiles of h: B-operand lane holds col nt*16+row16, K = kbase+0..7, 16+kbase+0..7
        #pragma unroll
        for (int nt = 0; nt < 8; ++nt) {
            const _Float16* bp = hT + (size_t)(nt * 16 + row16) * NN + j0 + kbase;
            v8h b0 = *(const v8h*)(bp);
            v8h b1 = *(const v8h*)(bp + 16);
            v16h Bh = __builtin_shufflevector(b0, b1,
                        0,1,2,3,4,5,6,7,8,9,10,11,12,13,14,15);
            acc[nt] = __builtin_amdgcn_wmma_f32_16x16x32_f16(
                          false, Ah, false, Bh, (short)0, acc[nt], false, false);
        }
    }

    // epilogue: divide by row sum, ELU, store
    const float linv = 1.f / l;
    float li[8];
    #pragma unroll
    for (int k = 0; k < 8; ++k) li[k] = __shfl(linv, kbase + k, 32);
    #pragma unroll
    for (int nt = 0; nt < 8; ++nt) {
        #pragma unroll
        for (int k = 0; k < 8; ++k) {
            float v = acc[nt][k] * li[k];
            v = v > 0.f ? v : (__expf(v) - 1.f);      // elu
            out[(size_t)(i0 + kbase + k) * DOUT + nt * 16 + row16] = v;
        }
    }
}

// ---------------------------------------------------------------------------
extern "C" void kernel_launch(void* const* d_in, const int* in_sizes, int n_in,
                              void* d_out, int out_size, void* d_ws, size_t ws_size,
                              hipStream_t stream)
{
    const float* input = (const float*)d_in[0];   // [8192,128] f32
    const int*   adj   = (const int*)d_in[1];     // [8192,8192] i32
    const float* W     = (const float*)d_in[2];   // [128,128] f32
    const float* a     = (const float*)d_in[3];   // [256,1] f32
    float*       outp  = (float*)d_out;           // [8192,128] f32

    _Float16* hT = (_Float16*)d_ws;                               // 2 MB
    float*    f1 = (float*)((char*)d_ws + (size_t)DOUT * NN * sizeof(_Float16));
    float*    f2 = f1 + NN;

    gat_proj<<<NN / 32, 128, 0, stream>>>(input, W, a, hT, f1, f2);
    gat_attn<<<NN / 16 / 4, 128, 0, stream>>>(adj, hT, f1, f2, outp);
}
